// MultiHeadAttention_6339371729116
// MI455X (gfx1250) — matrix operand order repaired and checked by reading, not compiled, not run on hardware
//
#include <hip/hip_runtime.h>

// ---------------------------------------------------------------------------
// MI455X (gfx1250) GQA attention block, bf16 WMMA pipeline + TDM staging.
//   Stage 1: f32 -> bf16 casts (x, Wq, Wk, Wv, Wo)
//   Stage 2: Q = x Wq^T, K = x Wk^T, V^T = (x Wv^T)^T   (bf16 WMMA GEMMs,
//            B tile staged in LDS by the Tensor Data Mover, double buffered)
//   Stage 3: causal flash attention per (head, 16-row q tile), f32 online softmax
//   Stage 4: out = Y Wo^T + bo  (f32 store)
// ---------------------------------------------------------------------------

typedef __bf16 bf16_t;
typedef __attribute__((ext_vector_type(8)))  __bf16 v8bf;
typedef __attribute__((ext_vector_type(16))) __bf16 v16bf;
typedef __attribute__((ext_vector_type(8)))  float  v8f;
typedef __attribute__((ext_vector_type(4)))  unsigned int v4u;
typedef __attribute__((ext_vector_type(8)))  int v8i;
typedef __attribute__((ext_vector_type(4)))  int v4i;

#define D_MODEL 2048
#define SEQ     2048
#define KV_DIM  512
#define DK      128
#define N_HEADS 16

#if __has_builtin(__builtin_amdgcn_tensor_load_to_lds) && __has_builtin(__builtin_amdgcn_s_wait_tensorcnt)
#define USE_TDM 1
#else
#define USE_TDM 0
#endif

__device__ __forceinline__ v16bf combine16(v8bf lo, v8bf hi) {
    return __builtin_shufflevector(lo, hi, 0,1,2,3,4,5,6,7,8,9,10,11,12,13,14,15);
}

// A fragment (16x32 bf16): lane holds row m = lane&15; halves 0..15 -> k=2*klo+0..15,
// klo = 8*(lane>=16).  Two contiguous 16B loads.
__device__ __forceinline__ v16bf load_frag_a(const bf16_t* row_k0, int klo) {
    v8bf lo = *(const v8bf*)(row_k0 + 2 * klo);
    v8bf hi = *(const v8bf*)(row_k0 + 2 * klo + 8);
    return combine16(lo, hi);
}

// B fragment (32x16 bf16): lane holds col n = lane&15, k = kbb + h, kbb=16*(lane>=16).
// Pointer pre-offset so the 16 halves are contiguous.
__device__ __forceinline__ v16bf load_frag_b(const bf16_t* p) {
    v8bf lo = *(const v8bf*)(p);
    v8bf hi = *(const v8bf*)(p + 8);
    return combine16(lo, hi);
}

__device__ __forceinline__ v8f wmma_bf16(v16bf a, v16bf b, v8f c) {
    return __builtin_amdgcn_wmma_f32_16x16x32_bf16(false, a, false, b, (short)0, c,
                                                   false, false);
}

// ---------------------------------------------------------------------------
// Stage 1: f32 -> bf16 cast
// ---------------------------------------------------------------------------
__global__ void cast_f32_bf16_kernel(const float* __restrict__ src,
                                     bf16_t* __restrict__ dst, int n) {
    int i = blockIdx.x * blockDim.x + threadIdx.x;
    int stride = gridDim.x * blockDim.x;
    for (; i < n; i += stride) dst[i] = (bf16_t)src[i];
}

// ---------------------------------------------------------------------------
// Stage 2/4 GEMM: C[M,N] = A[M,K] @ W[N,K]^T + bias.  A,W bf16 row-major.
// Block = 4 waves; block tile 128M x 64N, each wave owns 32M x 64N.
// B tile (64 rows x 32 k, 4KB) is staged into LDS -- via the Tensor Data Mover
// (double buffered, wave 0 issues, TENSORcnt-tracked) when available, else a
// cooperative copy.  LDS row pitch 80B: 16B aligned and bank-conflict-free.
// ---------------------------------------------------------------------------
#define BN 64
#define BK 32
#define LDS_ROW 40   // halves per row: 32 data + 8 pad  (80 bytes)

#if USE_TDM
// Issue one TDM 2D tile load: 64 rows x 64 bytes from W[n0..n0+63][k0..k0+31]
// into LDS at lds_off, with 16B padding after every 64B row (pitch 80B).
// D# fields per CDNA5 ISA ch.8: data_size=8B, tile=8x64, stride=K/4 elems,
// pad_interval=3 (16 dwords), pad_amount=3 (4 dwords), type=2.
// This toolchain's builtin takes 6 args (g0 v4u, g1 v8i, g2 v4i, g3 v4i,
// g4 v8i, cpol) -- 2D tile needs only groups 0/1, rest zero-filled.
__device__ __forceinline__ void tdm_load_b_tile(const bf16_t* gsrc,
                                                unsigned lds_off, int K) {
    unsigned long long ga = (unsigned long long)(uintptr_t)gsrc;
    unsigned td0 = (unsigned)(K >> 2);  // row length in 8-byte elements
    v4u g0 = { 1u,                                   // count=1 (valid), user mode
               lds_off,                              // lds_addr (bytes)
               (unsigned)ga,                         // global_addr[31:0]
               (unsigned)((ga >> 32) & 0x1FFFFFFull) | 0x80000000u }; // [56:32] | type=2
    v8i g1 = { (int)((3u << 16) | (1u << 20) | (3u << 22) | (3u << 25)),
               //    data_size=8B   pad_en      pad_int=16dw  pad_amt=4dw
               (int)((td0 & 0xFFFFu) << 16),         // tensor_dim0[15:0]
               (int)(((td0 >> 16) & 0xFFFFu) | (64u << 16)), // dim0 hi | tensor_dim1=64
               (int)(8u << 16),                      // tile_dim0 = 8 elems (64B)
               (int)64,                              // tile_dim1 = 64 rows
               (int)td0,                             // tensor_dim0_stride lo
               0, 0 };
    v4i gz4 = {0, 0, 0, 0};
    v8i gz8 = {0, 0, 0, 0, 0, 0, 0, 0};
    __builtin_amdgcn_tensor_load_to_lds(g0, g1, gz4, gz4, gz8, 0);
}
#endif

__global__ void gemm_bf16_wmma_kernel(const bf16_t* __restrict__ A,
                                      const bf16_t* __restrict__ W,
                                      const float* __restrict__ bias,
                                      void* __restrict__ Cout,
                                      int M, int N, int K,
                                      int store_f32, int trans_out) {
    __shared__ __align__(16) bf16_t lds_b[2][BN * LDS_ROW];   // 2 x 5 KB

    const int lane = threadIdx.x & 31;
    const int wave = threadIdx.x >> 5;
    const int nl   = lane & 15;
    const int hi8  = (lane >> 4) * 8;
    const int klo  = (lane >> 4) * 8;
    const int kbb  = (lane >> 4) * 16;

    const int n0 = blockIdx.x * BN;
    const int m0 = blockIdx.y * 128 + wave * 32;

    v8f c[8];
#pragma unroll
    for (int i = 0; i < 8; ++i) c[i] = (v8f){};

    const bf16_t* arow0 = A + (size_t)(m0 + nl) * K;
    const bf16_t* arow1 = A + (size_t)(m0 + 16 + nl) * K;
    const bf16_t* wtile = W + (size_t)n0 * K;

#if USE_TDM
    const unsigned lds_off0 = (unsigned)(uintptr_t)&lds_b[0][0];
    const unsigned lds_off1 = (unsigned)(uintptr_t)&lds_b[1][0];
    if (wave == 0) tdm_load_b_tile(wtile, lds_off0, K);   // prologue: buffer 0
#endif

    const int nsteps = K / BK;
    for (int s = 0; s < nsteps; ++s) {
        const int k0 = s * BK;
#if USE_TDM
        const bf16_t* bbuf = lds_b[s & 1];
        __syncthreads();                       // all waves done reading buf (s+1)&1
        if (wave == 0) {
            if (s + 1 < nsteps) {              // prefetch next tile into other buffer
                tdm_load_b_tile(wtile + (size_t)(k0 + BK),
                                ((s + 1) & 1) ? lds_off1 : lds_off0, K);
                __builtin_amdgcn_s_wait_tensorcnt(1);  // oldest (current buf) done
            } else {
                __builtin_amdgcn_s_wait_tensorcnt(0);
            }
        }
        __syncthreads();                       // current buffer visible to all waves
#else
        const bf16_t* bbuf = lds_b[0];
        __syncthreads();
        {   // cooperative copy: 64 rows x 64B, 128 threads x 32B
            int n = threadIdx.x >> 1, part = threadIdx.x & 1;
            const bf16_t* src = wtile + (size_t)n * K + k0 + part * 16;
            v8bf x0 = *(const v8bf*)(src);
            v8bf x1 = *(const v8bf*)(src + 8);
            *(v8bf*)(&lds_b[0][n * LDS_ROW + part * 16])     = x0;
            *(v8bf*)(&lds_b[0][n * LDS_ROW + part * 16 + 8]) = x1;
        }
        __syncthreads();
#endif
        v16bf a0 = load_frag_a(arow0 + k0, klo);
        v16bf a1 = load_frag_a(arow1 + k0, klo);
#pragma unroll
        for (int nt = 0; nt < 4; ++nt) {
            v16bf b = load_frag_b(bbuf + (nt * 16 + nl) * LDS_ROW + kbb);
            c[nt]     = wmma_bf16(a0, b, c[nt]);
            c[4 + nt] = wmma_bf16(a1, b, c[4 + nt]);
        }
    }

#pragma unroll
    for (int ms = 0; ms < 2; ++ms) {
#pragma unroll
        for (int nt = 0; nt < 4; ++nt) {
            v8f acc = c[ms * 4 + nt];
            int nn = n0 + nt * 16 + nl;
            float b = bias[nn];
#pragma unroll
            for (int r = 0; r < 8; ++r) {
                int mm = m0 + ms * 16 + r + hi8;
                float val = acc[r] + b;
                size_t idx = trans_out ? ((size_t)nn * M + mm) : ((size_t)mm * N + nn);
                if (store_f32) ((float*)Cout)[idx] = val;
                else           ((bf16_t*)Cout)[idx] = (bf16_t)val;
            }
        }
    }
}

// ---------------------------------------------------------------------------
// Stage 3: causal flash attention.  One wave per (head, 16-row query tile).
//   Q : bf16 [SEQ][D_MODEL]   (head h -> cols h*128 ..)
//   K : bf16 [SEQ][KV_DIM]    (kv head -> cols (h/4)*128 ..)
//   Vt: bf16 [KV_DIM][SEQ]    (transposed V -> B-operand contiguous in k)
//   Y : bf16 [SEQ][D_MODEL]
// ---------------------------------------------------------------------------
__global__ void attn_flash_wmma_kernel(const bf16_t* __restrict__ Q,
                                       const bf16_t* __restrict__ Kt,
                                       const bf16_t* __restrict__ Vt,
                                       bf16_t* __restrict__ Y) {
    __shared__ __align__(16) bf16_t lds_p[16 * 32];   // P tile, C-layout -> A-layout hop

    const int lane = threadIdx.x;
    const int nl   = lane & 15;
    const int hi8  = (lane >> 4) * 8;
    const int klo  = (lane >> 4) * 8;
    const int kbb  = (lane >> 4) * 16;

    const int t  = blockIdx.x;
    const int h  = blockIdx.y;
    const int kv = h >> 2;               // GROUP = 4
    const int q0 = t * 16;
    const int qlast = q0 + 15;
    const float scale = 0.08838834764831843f;  // 1/sqrt(128)

    const bf16_t* qbase = Q + (size_t)(q0 + nl) * D_MODEL + h * DK;
    v16bf qa[4];
#pragma unroll
    for (int kc = 0; kc < 4; ++kc) qa[kc] = load_frag_a(qbase + kc * 32, klo);

    v8f o[8];
#pragma unroll
    for (int dc = 0; dc < 8; ++dc) o[dc] = (v8f){};
    float mrun[8], lrun[8];
#pragma unroll
    for (int r = 0; r < 8; ++r) { mrun[r] = -__builtin_inff(); lrun[r] = 0.f; }

    const int jmax = qlast >> 5;
    for (int j = 0; j <= jmax; ++j) {
        const int kb0 = j << 5;
        const bool do1 = (kb0 + 16) <= qlast;

        v8f sc0 = {}, sc1 = {};
        const bf16_t* kr0 = Kt + (size_t)(kb0 + nl) * KV_DIM + kv * DK + kbb;
#pragma unroll
        for (int kc = 0; kc < 4; ++kc) sc0 = wmma_bf16(qa[kc], load_frag_b(kr0 + kc * 32), sc0);
        if (do1) {
            const bf16_t* kr1 = Kt + (size_t)(kb0 + 16 + nl) * KV_DIM + kv * DK + kbb;
#pragma unroll
            for (int kc = 0; kc < 4; ++kc) sc1 = wmma_bf16(qa[kc], load_frag_b(kr1 + kc * 32), sc1);
        }

#pragma unroll
        for (int r = 0; r < 8; ++r) {
            int qg = q0 + r + hi8;
            sc0[r] = ((kb0 + nl) <= qg) ? sc0[r] * scale : -__builtin_inff();
            if (do1)
                sc1[r] = ((kb0 + 16 + nl) <= qg) ? sc1[r] * scale : -__builtin_inff();
        }

        float corr[8];
#pragma unroll
        for (int r = 0; r < 8; ++r) {
            float v = sc0[r];
            if (do1) v = fmaxf(v, sc1[r]);
            v = fmaxf(v, __shfl_xor(v, 8, 16));
            v = fmaxf(v, __shfl_xor(v, 4, 16));
            v = fmaxf(v, __shfl_xor(v, 2, 16));
            v = fmaxf(v, __shfl_xor(v, 1, 16));
            float mn = fmaxf(mrun[r], v);
            corr[r] = __expf(mrun[r] - mn);
            mrun[r] = mn;

            float p0 = __expf(sc0[r] - mn);
            float p1 = do1 ? __expf(sc1[r] - mn) : 0.f;
            float rs = p0 + p1;
            rs += __shfl_xor(rs, 8, 16);
            rs += __shfl_xor(rs, 4, 16);
            rs += __shfl_xor(rs, 2, 16);
            rs += __shfl_xor(rs, 1, 16);
            lrun[r] = lrun[r] * corr[r] + rs;

            int m = r + hi8;
            lds_p[m * 32 + nl]      = (bf16_t)p0;
            lds_p[m * 32 + 16 + nl] = (bf16_t)p1;
        }
        __syncthreads();

#pragma unroll
        for (int dc = 0; dc < 8; ++dc)
#pragma unroll
            for (int r = 0; r < 8; ++r) o[dc][r] *= corr[r];

        v8bf plo = *(const v8bf*)(lds_p + nl * 32 + 2 * klo);
        v8bf phi = *(const v8bf*)(lds_p + nl * 32 + 2 * klo + 8);
        v16bf pa = combine16(plo, phi);

#pragma unroll
        for (int dc = 0; dc < 8; ++dc) {
            const bf16_t* vrow = Vt + (size_t)(kv * DK + dc * 16 + nl) * SEQ + kb0 + kbb;
            o[dc] = wmma_bf16(pa, load_frag_b(vrow), o[dc]);
        }
        __syncthreads();
    }

#pragma unroll
    for (int dc = 0; dc < 8; ++dc) {
#pragma unroll
        for (int r = 0; r < 8; ++r) {
            float val = o[dc][r] / lrun[r];
            Y[(size_t)(q0 + r + hi8) * D_MODEL + h * DK + dc * 16 + nl] = (bf16_t)val;
        }
    }
}

// ---------------------------------------------------------------------------
// Host-side launch
// ---------------------------------------------------------------------------
extern "C" void kernel_launch(void* const* d_in, const int* in_sizes, int n_in,
                              void* d_out, int out_size, void* d_ws, size_t ws_size,
                              hipStream_t stream) {
    const float* x  = (const float*)d_in[0];
    const float* Wq = (const float*)d_in[1];
    const float* bq = (const float*)d_in[2];
    const float* Wk = (const float*)d_in[3];
    const float* bk = (const float*)d_in[4];
    const float* Wv = (const float*)d_in[5];
    const float* bv = (const float*)d_in[6];
    const float* Wo = (const float*)d_in[7];
    const float* bo = (const float*)d_in[8];
    float* out = (float*)d_out;

    char* ws = (char*)d_ws;
    size_t off = 0;
    auto carve = [&](size_t elems) { bf16_t* p = (bf16_t*)(ws + off); off += elems * sizeof(bf16_t); return p; };
    bf16_t* xh  = carve((size_t)SEQ * D_MODEL);
    bf16_t* Wqh = carve((size_t)D_MODEL * D_MODEL);
    bf16_t* Wkh = carve((size_t)KV_DIM * D_MODEL);
    bf16_t* Wvh = carve((size_t)KV_DIM * D_MODEL);
    bf16_t* Woh = carve((size_t)D_MODEL * D_MODEL);
    bf16_t* Qh  = carve((size_t)SEQ * D_MODEL);
    bf16_t* Kh  = carve((size_t)SEQ * KV_DIM);
    bf16_t* Vt  = carve((size_t)KV_DIM * SEQ);
    bf16_t* Yh  = carve((size_t)SEQ * D_MODEL);
    (void)ws_size; (void)n_in; (void)in_sizes; (void)out_size;

    auto cast = [&](const float* s, bf16_t* d, int n) {
        int blocks = (n + 255) / 256;
        if (blocks > 8192) blocks = 8192;
        cast_f32_bf16_kernel<<<blocks, 256, 0, stream>>>(s, d, n);
    };
    cast(x,  xh,  SEQ * D_MODEL);
    cast(Wq, Wqh, D_MODEL * D_MODEL);
    cast(Wk, Wkh, KV_DIM * D_MODEL);
    cast(Wv, Wvh, KV_DIM * D_MODEL);
    cast(Wo, Woh, D_MODEL * D_MODEL);

    // Q = x Wq^T + bq -> bf16 [SEQ][D_MODEL]
    gemm_bf16_wmma_kernel<<<dim3(D_MODEL / 64, SEQ / 128), 128, 0, stream>>>(
        xh, Wqh, bq, Qh, SEQ, D_MODEL, D_MODEL, /*f32=*/0, /*trans=*/0);
    // K = x Wk^T + bk -> bf16 [SEQ][KV_DIM]
    gemm_bf16_wmma_kernel<<<dim3(KV_DIM / 64, SEQ / 128), 128, 0, stream>>>(
        xh, Wkh, bk, Kh, SEQ, KV_DIM, D_MODEL, 0, 0);
    // V^T = (x Wv^T + bv)^T -> bf16 [KV_DIM][SEQ]
    gemm_bf16_wmma_kernel<<<dim3(KV_DIM / 64, SEQ / 128), 128, 0, stream>>>(
        xh, Wvh, bv, Vt, SEQ, KV_DIM, D_MODEL, 0, /*trans=*/1);

    // flash attention
    attn_flash_wmma_kernel<<<dim3(SEQ / 16, N_HEADS), 32, 0, stream>>>(Qh, Kh, Vt, Yh);

    // out = Y Wo^T + bo -> f32
    gemm_bf16_wmma_kernel<<<dim3(D_MODEL / 64, SEQ / 128), 128, 0, stream>>>(
        Yh, Woh, bo, out, SEQ, D_MODEL, D_MODEL, /*f32=*/1, 0);
}
